// EncoderPairFeatures_37967510897119
// MI455X (gfx1250) — matrix operand order repaired and checked by reading, not compiled, not run on hardware
//
#include <hip/hip_runtime.h>
#include <stdint.h>

#define NRES 768
#define FIN  316
#define KPAD 320
#define DP   128
#define NKK  10      // KPAD / 32

typedef __attribute__((ext_vector_type(16))) __bf16 v16bf;
typedef __attribute__((ext_vector_type(8)))  float  v8f;
typedef __attribute__((ext_vector_type(4)))  float  f4;

// ---------- small helpers ----------

__device__ inline uint32_t bf16bits(float f) {
  union { __bf16 h; uint16_t u; } c; c.h = (__bf16)f; return (uint32_t)c.u;
}

// searchsorted(limits, v) for uniform limits: count of limits strictly < v
__device__ inline int bin_idx(float v, float lo, float invstep, int nlim) {
  if (!(v > lo)) return 0;
  int i = (int)floorf((v - lo) * invstep) + 1;
  return (i > nlim) ? nlim : i;
}

struct V3 { float x, y, z; };
__device__ inline V3 vsub(V3 a, V3 b) { return {a.x-b.x, a.y-b.y, a.z-b.z}; }
__device__ inline float vdot(V3 a, V3 b) { return a.x*b.x + a.y*b.y + a.z*b.z; }
__device__ inline V3 vcross(V3 a, V3 b) {
  return {a.y*b.z - a.z*b.y, a.z*b.x - a.x*b.z, a.x*b.y - a.y*b.x};
}
__device__ inline V3 vnorm(V3 v) {
  float n = fmaxf(sqrtf(vdot(v, v) + 1e-16f), 1e-8f);
  float r = 1.0f / n;
  return {v.x*r, v.y*r, v.z*r};
}
__device__ inline V3 ld3(const float* p) { return {p[0], p[1], p[2]}; }

__device__ inline float bond_angle(V3 a, V3 b, V3 c) {
  V3 u = vnorm(vsub(b, a));
  V3 v = vnorm(vsub(c, a));
  V3 cr = vcross(u, v);
  return atan2f(sqrtf(vdot(cr, cr) + 1e-16f), vdot(u, v));
}

__device__ inline float dihedral(V3 a, V3 b, V3 c, V3 d) {
  V3 b1 = vsub(b, a), b2 = vsub(c, b), b3 = vsub(d, c);
  V3 n1 = vnorm(vcross(b1, b2));
  V3 n2 = vnorm(vcross(b2, b3));
  V3 m  = vcross(n1, n2);
  float t = vdot(m, b2);
  float s = (t > 0.f) ? 1.f : ((t < 0.f) ? -1.f : 0.f);
  return atan2f(s * sqrtf(vdot(m, m) + 1e-16f), vdot(n1, n2));
}

// CDNA5 async global->LDS copy (ASYNCcnt path)
__device__ inline void async_load_b128(void* lds_dst, const void* gsrc) {
  uint32_t lo = (uint32_t)(uintptr_t)lds_dst;  // low 32 bits of flat addr == LDS byte offset
  asm volatile("global_load_async_to_lds_b128 %0, %1, off"
               :: "v"(lo), "v"(gsrc) : "memory");
}

// ---------- kernel 0: pack W (f32 [128][316]) into pre-swizzled bf16 B fragments ----------
// layout: dword offset o = ((w*32 + l)*NKK + kk)*8 + v
//   K = kk*32 + (l>=16 ? 16 : 0) + 2v ; N = w*16 + (l&15)
//   dword = { bf16(W[N][K]) , bf16(W[N][K+1]) }   (zero beyond FIN)
__global__ __launch_bounds__(256) void pack_w(const float* __restrict__ W,
                                              uint32_t* __restrict__ Bg) {
  int o = blockIdx.x * 256 + threadIdx.x;
  if (o >= 8 * 32 * NKK * 8) return;
  int v   = o & 7;
  int rem = o >> 3;
  int kk  = rem % NKK;
  int rem2 = rem / NKK;
  int l   = rem2 & 31;
  int w   = rem2 >> 5;
  int K = kk * 32 + ((l >= 16) ? 16 : 0) + 2 * v;
  int N = w * 16 + (l & 15);
  float flo = (K     < FIN) ? W[N * FIN + K    ] : 0.f;
  float fhi = (K + 1 < FIN) ? W[N * FIN + K + 1] : 0.f;
  Bg[o] = bf16bits(flo) | (bf16bits(fhi) << 16);
}

// ---------- main kernel: one block per residue row i ----------
__global__ __launch_bounds__(256) void pairfeat_kernel(
    const float* __restrict__ mask, const float* __restrict__ coord_mask,
    const float* __restrict__ coords_nm, const float* __restrict__ coords,
    const float* __restrict__ bias, const float* __restrict__ gamma,
    const float* __restrict__ beta, const uint32_t* __restrict__ Bg,
    float* __restrict__ out)
{
  __shared__ uint32_t Atile[16 * KPAD / 2];                 // 16 x 320 bf16 (10 KB)
  __shared__ float ytile[16][DP + 2];                       // 16 x 130 f32
  __shared__ __align__(16) float biasS[DP];
  __shared__ __align__(16) float gammaS[DP];
  __shared__ __align__(16) float betaS[DP];

  const int i   = blockIdx.x;
  const int tid = threadIdx.x;
  const int wv  = tid >> 5;   // wave id = d-block
  const int ln  = tid & 31;

  // --- stage bias / gamma / beta via async global->LDS b128 ---
  if (tid < 96) {
    int grp = tid >> 5;
    int e4  = tid & 31;
    const float* src = (grp == 0) ? bias : ((grp == 1) ? gamma : beta);
    float*       dst = (grp == 0) ? biasS : ((grp == 1) ? gammaS : betaS);
    async_load_b128(&dst[e4 * 4], &src[e4 * 4]);
  }
  asm volatile("s_wait_asynccnt 0" ::: "memory");
  __syncthreads();

  // --- persistent B fragments (wave's 320x16 bf16 slice, 80 VGPRs) ---
  v16bf Bf[NKK];
  {
    const uint32_t* bp = Bg + (size_t)(wv * 32 + ln) * (NKK * 8);
#pragma unroll
    for (int kk = 0; kk < NKK; ++kk)
      Bf[kk] = *(const v16bf*)(bp + kk * 8);
  }

  // --- per-row-i constants ---
  const float mi   = mask[i];
  const float hcbi = coord_mask[i * 4 + 3];
  const V3 CAnm = ld3(coords_nm + (i * 4 + 1) * 3);
  const V3 Niv  = ld3(coords    + (i * 4 + 0) * 3);
  const V3 CAiv = ld3(coords    + (i * 4 + 1) * 3);
  const V3 CBiv = ld3(coords    + (i * 4 + 3) * 3);

  const float PI = 3.14159265358979323846f;
  const float angInv = 19.0f / (2.0f * PI);

  for (int jt = 0; jt < 48; ++jt) {
    const int j0 = jt * 16;

    // zero the A tile (2560 dwords)
#pragma unroll
    for (int q = 0; q < 10; ++q) Atile[q * 256 + tid] = 0u;
    __syncthreads();

    // threads 0..15: compute the 10 (index, weight) pairs for pair (i, j0+t)
    if (tid < 16) {
      const int j = j0 + tid;
      const float mj   = mask[j];
      const float pm   = mi * mj;
      const float pm2  = pm * pm;
      const float hcbj = coord_mask[j * 4 + 3];
      __bf16* Ar = ((__bf16*)Atile) + tid * KPAD;

      // seq separation (127 bins)
      Ar[bin_idx((float)(i - j), -62.5f, 1.0f, 126)] = (__bf16)pm;

      // CA(i) - atom(j) distances, 4 x 21 bins
#pragma unroll
      for (int a = 0; a < 4; ++a) {
        V3 p  = ld3(coords_nm + (j * 4 + a) * 3);
        V3 dv = vsub(CAnm, p);
        float d = sqrtf(vdot(dv, dv) + 1e-10f);
        float wgt = pm2 * ((a == 3) ? hcbj : 1.f);
        Ar[127 + a * 21 + bin_idx(d, 0.1f, 10.0f, 20)] = (__bf16)wgt;
      }

      // orientation angles, 5 x 21 bins
      V3 Njv  = ld3(coords + (j * 4 + 0) * 3);
      V3 CAjv = ld3(coords + (j * 4 + 1) * 3);
      V3 CBjv = ld3(coords + (j * 4 + 3) * 3);
      float ang[5];
      ang[0] = dihedral(Niv,  CAiv, CBiv, CBjv);
      ang[1] = dihedral(Njv,  CAjv, CBjv, CBiv);
      ang[2] = bond_angle(CAiv, CBiv, CBjv);
      ang[3] = bond_angle(CAjv, CBjv, CBiv);
      ang[4] = dihedral(CAiv, CBiv, CBjv, CAjv);
      const float wor = pm2 * hcbi * hcbi;   // matches reference (has_cb_i twice)
#pragma unroll
      for (int a = 0; a < 5; ++a)
        Ar[211 + a * 21 + bin_idx(ang[a], -PI, angInv, 20)] = (__bf16)wor;
    }
    __syncthreads();

    // --- WMMA: 16 pairs x 16 d, K = 320 in 10 steps ---
    v8f acc = {0.f, 0.f, 0.f, 0.f, 0.f, 0.f, 0.f, 0.f};
    const int am = ln & 15;
    const int ahi = (ln >= 16) ? 16 : 0;
#pragma unroll
    for (int kk = 0; kk < NKK; ++kk) {
      union { v16bf v; uint32_t u[8]; } af;
#pragma unroll
      for (int q = 0; q < 8; ++q) {
        int k = kk * 32 + 2 * q + ahi;
        af.u[q] = Atile[(am * KPAD + k) >> 1];
      }
      acc = __builtin_amdgcn_wmma_f32_16x16x32_bf16(
          false, af.v, false, Bf[kk], (short)0, acc, false, false);
    }

    // --- epilogue: + bias into LDS y tile ---
    {
      const int d = wv * 16 + (ln & 15);
      const int mrow = (ln >> 4) * 8;
#pragma unroll
      for (int r = 0; r < 8; ++r)
        ytile[mrow + r][d] = acc[r] + biasS[d];
    }
    __syncthreads();

    // --- LayerNorm over d=128, 16 lanes per pair ---
    {
      const int p   = tid >> 4;   // pair within tile
      const int seg = tid & 15;   // 8 consecutive d each
      float vals[8];
      float s1 = 0.f, s2 = 0.f;
#pragma unroll
      for (int q = 0; q < 8; ++q) {
        float y = ytile[p][seg * 8 + q];
        vals[q] = y; s1 += y; s2 += y * y;
      }
#pragma unroll
      for (int msk = 1; msk < 16; msk <<= 1) {
        s1 += __shfl_xor(s1, msk, 32);
        s2 += __shfl_xor(s2, msk, 32);
      }
      float mu   = s1 * (1.0f / 128.0f);
      float var  = s2 * (1.0f / 128.0f) - mu * mu;
      float rstd = rsqrtf(var + 1e-5f);
      const int j = j0 + p;
      const float pm = mi * mask[j];
      f4 o0, o1;
#pragma unroll
      for (int q = 0; q < 8; ++q) {
        int d = seg * 8 + q;
        float y = ((vals[q] - mu) * rstd * gammaS[d] + betaS[d]) * pm;
        if (q < 4) o0[q] = y; else o1[q - 4] = y;
      }
      float* op = out + ((size_t)(i * NRES + j)) * DP + seg * 8;
      *(f4*)op = o0;
      *(f4*)(op + 4) = o1;
    }
    __syncthreads();
  }
}

// ---------- launch ----------
extern "C" void kernel_launch(void* const* d_in, const int* in_sizes, int n_in,
                              void* d_out, int out_size, void* d_ws, size_t ws_size,
                              hipStream_t stream) {
  const float* mask       = (const float*)d_in[0];
  const float* coord_mask = (const float*)d_in[1];
  const float* coords_nm  = (const float*)d_in[2];
  const float* coords     = (const float*)d_in[3];
  const float* W          = (const float*)d_in[4];
  const float* b          = (const float*)d_in[5];
  const float* gamma      = (const float*)d_in[6];
  const float* beta       = (const float*)d_in[7];
  uint32_t* Bg = (uint32_t*)d_ws;      // 8*32*10*8 dwords = 80 KB
  float* out = (float*)d_out;

  pack_w<<<80, 256, 0, stream>>>(W, Bg);
  pairfeat_kernel<<<NRES, 256, 0, stream>>>(mask, coord_mask, coords_nm, coords,
                                            b, gamma, beta, Bg, out);
}
